// SPEClassifier_47115791237571
// MI455X (gfx1250) — matrix operand order repaired and checked by reading, not compiled, not run on hardware
//
#include <hip/hip_runtime.h>
#include <hip/hip_bf16.h>
#include <math.h>

// Sizes fixed by the reference
#define Dd     256
#define Kk     512
#define Ll     64
#define Bb     8
#define Ss     16
#define Cc     64
#define NROW   512           // B*L
#define F_LOG2PI 1.8378770664093453f
#define F_LOGS   2.7725887222397811f   // log(16)

typedef __attribute__((ext_vector_type(16))) _Float16 v16h;
typedef __attribute__((ext_vector_type(2)))  _Float16 h2;
typedef __attribute__((ext_vector_type(8)))  float    v8f;

// ---------------- counter-based RNG (PCG hash -> Box-Muller) ----------------
__device__ __forceinline__ unsigned pcg_hash(unsigned x) {
    x = x * 747796405u + 2891336453u;
    unsigned w = ((x >> ((x >> 28u) + 4u)) ^ x) * 277803737u;
    return (w >> 22u) ^ w;
}
__device__ __forceinline__ float gauss_at(unsigned idx) {
    unsigned h1 = pcg_hash(2u * idx + 1u);
    unsigned h2v = pcg_hash(2u * idx + 2u);
    float u1 = (float)(h1 >> 8) * (1.0f / 16777216.0f) + (0.5f / 16777216.0f); // (0,1]
    float u2 = (float)(h2v >> 8) * (1.0f / 16777216.0f);
    return sqrtf(-2.0f * __logf(u1)) * __cosf(6.2831853071795865f * u2);
}

// ---------------- kernel 0: group rows by (rank, class), record order -------
__global__ void k_group(const float* __restrict__ params,
                        const int*   __restrict__ labels,
                        float* __restrict__ grouped,
                        int*   __restrict__ order) {
    __shared__ int sh_rc[2];
    const int i = blockIdx.x;
    if (threadIdx.x == 0) {
        int c = labels[i];
        int r = 0;
        for (int j = 0; j < i; ++j) r += (labels[j] == c) ? 1 : 0;
        sh_rc[0] = r; sh_rc[1] = c;
        order[r * Ll + c] = i;
    }
    __syncthreads();
    const int r = sh_rc[0], c = sh_rc[1];
    float* dst = grouped + (size_t)(r * Ll + c) * Kk;
    const float* src = params + (size_t)i * Kk;
    for (int t = threadIdx.x; t < Kk; t += blockDim.x) dst[t] = src[t];
}

// ---------------- kernel 1: prototype fusion per half ----------------------
// half 0: support = ranks 4..7 ; half 1: support = ranks 0..3
__global__ void k_proto(const float* __restrict__ grouped,
                        const float* __restrict__ heps,
                        float* __restrict__ pM,
                        float* __restrict__ pV,
                        float* __restrict__ pH) {
    const int g = blockIdx.x * blockDim.x + threadIdx.x;   // 2*64*256 threads
    const int half = g >> 14;
    const int c    = (g >> 8) & 63;
    const int d    = g & 255;
    const float eps_var = __expf(heps[0]);
    float invs = 0.0f, ms = 0.0f;
    #pragma unroll
    for (int j = 0; j < 4; ++j) {
        const int rr = (half == 0) ? (4 + j) : j;
        const float* p = grouped + (size_t)(rr * Ll + c) * Kk;
        const float m = p[d];
        const float h = p[Dd + d];
        const float iv = 1.0f / (eps_var + __expf(h));
        invs += iv;
        ms   += m * iv;
    }
    const float nv = 1.0f / invs;
    const float nm = nv * ms;
    const float ph = __logf(eps_var + nv);
    const int o = (half * Ll + c) * Dd + d;
    pM[o] = nm;
    pV[o] = eps_var + nv;      // exp(ph)
    pH[o] = ph;
}

// ---------------- kernel 2: fused MC-logpdf via WMMA + logsumexps ----------
// One workgroup per (half, query-rank, query-class). The 16x256 noise matrix
// (eps, eps^2 interleaved along K) is generated ONCE per workgroup into LDS,
// pre-packed in a contiguous-k A layout (k = 16*half + elem), so each wave's per-kb A
// fragment is two contiguous ds_load_b128s shared by all 4 waves.
__global__ void __launch_bounds__(128) k_logits(
        const float* __restrict__ grouped,
        const int*   __restrict__ order,
        const float* __restrict__ pM,
        const float* __restrict__ pV,
        const float* __restrict__ pH,
        float* __restrict__ out) {
    __shared__ unsigned afrag[16][32][8];   // [kb][lane][vgpr] = (eps,eps^2) f16x2 ; 16 KB
    __shared__ float qm[Dd];
    __shared__ float qv[Dd];
    __shared__ float tl[Ss][Ll];
    __shared__ float lse[Ss];
    __shared__ float Fsh;

    const int blk  = blockIdx.x;          // 0..511
    const int half = blk >> 8;
    const int br   = (blk >> 6) & 3;
    const int lq   = blk & 63;
    const int rq   = half * 4 + br;       // global query rank
    const int tid  = threadIdx.x;
    const int lane = tid & 31;
    const int wave = tid >> 5;
    const int lpLoc = lane & 15;
    const int grp   = lane >> 4;          // half-wave group
    const int lp    = wave * 16 + lpLoc;  // this lane's prototype column

    // ---- stage query mean / variance in LDS ----
    const float* q = grouped + (size_t)(rq * Ll + lq) * Kk;
    for (int d = tid; d < Dd; d += 128) {
        qm[d] = q[d];
        qv[d] = __expf(q[Dd + d]);
    }

    // ---- cooperative noise generation, packed in A-fragment layout ----
    // (s,d) -> kb = d>>4; dl = d&15; grpA = dl>>3; j = dl&7
    const unsigned rngBase = (unsigned)((rq * Ll + lq) * (Ss * Dd));
    for (int it = tid; it < Ss * Dd; it += 128) {     // 32 iters
        const int s  = it >> 8;
        const int d  = it & 255;
        const float e = gauss_at(rngBase + (unsigned)it);
        h2 p; p[0] = (_Float16)e; p[1] = (_Float16)(e * e);
        const int kb   = d >> 4;
        const int dl   = d & 15;
        const int grpA = dl >> 3;
        const int j    = dl & 7;
        afrag[kb][grpA * 16 + s][j] = __builtin_bit_cast(unsigned, p);
    }
    __syncthreads();

    const float* pMrow = pM + (size_t)(half * Ll + lp) * Dd;
    const float* pVrow = pV + (size_t)(half * Ll + lp) * Dd;
    const float* pHrow = pH + (size_t)(half * Ll + lp) * Dd;

    float kcPart = 0.0f, lmPart = 0.0f;
    v8f acc = {0.f, 0.f, 0.f, 0.f, 0.f, 0.f, 0.f, 0.f};

    for (int kb = 0; kb < 16; ++kb) {
        // ---- A fragment: 32B aligned LDS read (2x ds_load_b128) ----
        const v16h a = *(const v16h*)(&afrag[kb][lane][0]);
        // ---- B fragment: (c1, c2) pairs from product-Gaussian algebra ----
        v16h b;
        const int dB = kb * 16 + grp * 8;
        #pragma unroll
        for (int t = 0; t < 8; ++t) {
            const int d = dB + t;
            const float pm = pMrow[d];
            const float pv = pVrow[d];
            const float ph = pHrow[d];
            const float v1 = qv[d];
            const float dm = qm[d] - pm;
            const float vs = v1 + pv;
            const float inv = 1.0f / vs;
            const float sig = sqrtf(v1 * pv * inv);            // product sigma
            b[2 * t]     = (_Float16)(2.0f * dm * inv * sig);  // c1
            b[2 * t + 1] = (_Float16)(v1 * inv);               // c2
            kcPart += F_LOG2PI + ph + dm * dm * pv * inv * inv;
            lmPart += F_LOG2PI + __logf(vs) + dm * dm * inv;
        }
        acc = __builtin_amdgcn_wmma_f32_16x16x32_f16(
            /*neg_a=*/false, a, /*neg_b=*/false, b,
            /*c_mod=*/(short)0, acc, /*reuse_a=*/false, /*reuse_b=*/false);
    }

    // each lane covered half the d's for its lp; partner is lane^16
    const float kcFull = kcPart + __shfl_xor(kcPart, 16, 32);
    const float lmFull = lmPart + __shfl_xor(lmPart, 16, 32);

    // scatter tl[s][lp]: acc element r holds (M = r + 8*grp, N = lpLoc)
    #pragma unroll
    for (int rI = 0; rI < 8; ++rI) {
        const int s = rI + grp * 8;
        tl[s][lp] = -0.5f * (kcFull + acc[rI]);
    }
    __syncthreads();

    // logsumexp over prototypes, per sample
    if (tid < Ss) {
        float mx = -1e30f;
        for (int j = 0; j < Ll; ++j) mx = fmaxf(mx, tl[tid][j]);
        float sm = 0.0f;
        for (int j = 0; j < Ll; ++j) sm += __expf(tl[tid][j] - mx);
        lse[tid] = mx + __logf(sm);
    }
    __syncthreads();
    // logsumexp over samples of (-lse), minus log S
    if (tid == 0) {
        float mx = -1e30f;
        for (int s = 0; s < Ss; ++s) mx = fmaxf(mx, -lse[s]);
        float sm = 0.0f;
        for (int s = 0; s < Ss; ++s) sm += __expf(-lse[s] - mx);
        Fsh = mx + __logf(sm) - F_LOGS;
    }
    __syncthreads();

    // final logits row (label_map is identity: column == prototype class)
    if (grp == 0) {
        const int row = order[rq * Ll + lq];   // un-sort to input order
        out[(size_t)row * Cc + lp] = -0.5f * lmFull + Fsh;
    }
}

// ---------------------------------------------------------------------------
extern "C" void kernel_launch(void* const* d_in, const int* in_sizes, int n_in,
                              void* d_out, int out_size, void* d_ws, size_t ws_size,
                              hipStream_t stream) {
    const float* params = (const float*)d_in[0];   // (512, 512) f32
    const float* heps   = (const float*)d_in[1];   // scalar f32
    const int*   labels = (const int*)d_in[2];     // (512,) int
    float* out = (float*)d_out;                    // (512, 64) f32

    char* ws = (char*)d_ws;
    float* grouped = (float*)ws;                           // 8*64*512 f32 = 1 MB
    int*   order   = (int*)(ws + (size_t)1048576);         // 512 ints
    float* pM      = (float*)(ws + (size_t)1056768);       // 2*64*256 f32 each
    float* pV      = pM + 2 * Ll * Dd;
    float* pH      = pV + 2 * Ll * Dd;

    k_group <<<NROW, 256, 0, stream>>>(params, labels, grouped, order);
    k_proto <<<(2 * Ll * Dd) / 256, 256, 0, stream>>>(grouped, heps, pM, pV, pH);
    k_logits<<<NROW, 128, 0, stream>>>(grouped, order, pM, pV, pH, out);
}
